// MultiHeadCoAttention_71940702208210
// MI455X (gfx1250) — matrix operand order repaired and checked by reading, not compiled, not run on hardware
//
// MultiHeadCoAttention for MI455X (gfx1250), wave32 + WMMA bf16.
//
// Pipeline (all on `stream`, workspace-carved, no scores materialization):
//   1) fp32->bf16 converts (ctx1, ctx2, 6 weight matrices)
//   2) proj_gemm (WMMA bf16, 32x32 register-blocked, software-pipelined)
//      -> k1,v1,k2,v2 in [B,H,L,64] bf16
//   3) colstats : per (b,h,k) max/sumexp over q   (softmax axis=2)
//   4) headstats: per (b,q,k) max/sumexp over h   (softmax axis=1)
//   5) ctx12_av : O = softmax_q(S) @ v1 (recompute S tiles via WMMA)
//   6) ctx21_av : O = softmax_h(S)^T @ v2 (LDS transpose of P tile)
//   7) proj_gemm fp32 mode -> d_out (context_2_to_1 then context_1_to_2)
//
// Workspace use: ~205 MB (bf16 operands + 67MB head-softmax stats).

#include <hip/hip_runtime.h>
#include <cstddef>
#include <cstdint>

#define L1C 1024
#define L2C 1024
#define BC  8
#define D1C 512
#define D2C 1024
#define HC  16
#define MC  1024
#define DPHC 64
#define PSTR 40   // LDS row stride (elements): multiple of 8 (16B), odd*8 for bank spread

typedef __bf16 bf16_t;
typedef bf16_t v16bf __attribute__((ext_vector_type(16)));
typedef bf16_t v8bf  __attribute__((ext_vector_type(8)));
typedef float  v8f   __attribute__((ext_vector_type(8)));

__device__ __forceinline__ bf16_t f2bf(float f) {
  uint32_t u = __builtin_bit_cast(uint32_t, f);
  u += 0x7FFFu + ((u >> 16) & 1u);          // round-to-nearest-even
  uint16_t h = (uint16_t)(u >> 16);
  return __builtin_bit_cast(bf16_t, h);
}

__device__ __forceinline__ v8f zerov8() {
  v8f c = {0.f, 0.f, 0.f, 0.f, 0.f, 0.f, 0.f, 0.f};
  return c;
}

__device__ __forceinline__ v8f wmma_bf16(v16bf a, v16bf b, v8f c) {
  // D = A(16x32) * B(32x16) + C, fp32 accum
  return __builtin_amdgcn_wmma_f32_16x16x32_bf16(
      /*neg_a=*/false, a, /*neg_b=*/false, b,
      /*c_mod=*/(short)0, c, /*reuse_a=*/false, /*reuse_b=*/false);
}

// A-fragment (16x32, row-major source, row stride `ld` elements).
// Lane l: row = l&15; elems 0..7 -> K = kb+i, elems 8..15 -> K = kb+8+i, kb=(l>>4)*16.
__device__ __forceinline__ v16bf load_a(const bf16_t* __restrict__ base, int ld, int lane) {
  const int row = lane & 15, kb = (lane >> 4) * 16;
  const bf16_t* p = base + (size_t)row * ld + kb;
  v8bf a0 = *(const v8bf*)(p);
  v8bf a1 = *(const v8bf*)(p + 8);
  v16bf a;
#pragma unroll
  for (int i = 0; i < 8; ++i) { a[i] = a0[i]; a[8 + i] = a1[i]; }
  return a;
}

// B-fragment (32x16) where B[k][n] = Wrow[n][k] and Wrow is row-major [N x K].
// Lane l: col = l&15; elems i -> K = kb+i, kb=(l>>4)*16. One 32B contiguous load.
__device__ __forceinline__ v16bf load_bT(const bf16_t* __restrict__ base, int ld, int lane) {
  const int col = lane & 15, kb = (lane >> 4) * 16;
  return *(const v16bf*)(base + (size_t)col * ld + kb);
}

// ---------------------------------------------------------------- convert
__global__ void __launch_bounds__(256)
cvt_f32_bf16(const float* __restrict__ src, bf16_t* __restrict__ dst, int n) {
  int i = blockIdx.x * blockDim.x + threadIdx.x;
  const int stride = gridDim.x * blockDim.x;
  for (; i < n; i += stride) dst[i] = f2bf(src[i]);
}

// ---------------------------------------------------------------- GEMM: Y = X @ W^T + bias
// X: [R x K] bf16 row-major, W: [N x K] bf16 row-major, bias: [N] f32.
// Each wave owns a 32x32 output block: 2 A-frags x 2 B-frags -> 4 WMMAs per K-step
// (2x operand reuse), with next-K fragments preloaded so WMMAs overlap the loads.
// outF != nullptr: write fp32 [R x N] (final projections, R rows == [L,B] order).
// else: scatter bf16 into [B,H,L,64] head-major (input projections).
__global__ void __launch_bounds__(128)
proj_gemm(const bf16_t* __restrict__ X, const bf16_t* __restrict__ W,
          const float* __restrict__ bias, int K, int N, int Bdim, int Ldim,
          float* __restrict__ outF, bf16_t* __restrict__ outH) {
  const int lane = threadIdx.x & 31;
  const int tile = blockIdx.x * 4 + (threadIdx.x >> 5);
  const int ntiles = N >> 5;                // 32-wide column blocks
  const int tn = tile % ntiles, tr = tile / ntiles;
  const int col = lane & 15, rbase = (lane >> 4) * 8;

  const bf16_t* xa0 = X + (size_t)(tr * 32) * K;
  const bf16_t* xa1 = xa0 + (size_t)16 * K;
  const bf16_t* wb0 = W + (size_t)(tn * 32) * K;
  const bf16_t* wb1 = wb0 + (size_t)16 * K;

  v8f c00, c01, c10, c11;
  {
    const float bv0 = bias[tn * 32 + col];
    const float bv1 = bias[tn * 32 + 16 + col];
#pragma unroll
    for (int r = 0; r < 8; ++r) { c00[r] = bv0; c01[r] = bv1; c10[r] = bv0; c11[r] = bv1; }
  }

  // software pipeline: fragments for K-step 0
  v16bf a0 = load_a(xa0, K, lane);
  v16bf a1 = load_a(xa1, K, lane);
  v16bf b0 = load_bT(wb0, K, lane);
  v16bf b1 = load_bT(wb1, K, lane);

  for (int kw = 0; kw < K; kw += 32) {
    const int nk = kw + 32;
    v16bf na0, na1, nb0, nb1;
    const bool more = nk < K;
    if (more) {
      __builtin_prefetch(xa0 + nk + 256, 0, 1);   // global_prefetch_b8 (speculative)
      __builtin_prefetch(wb0 + nk + 256, 0, 1);
      na0 = load_a(xa0 + nk, K, lane);
      na1 = load_a(xa1 + nk, K, lane);
      nb0 = load_bT(wb0 + nk, K, lane);
      nb1 = load_bT(wb1 + nk, K, lane);
    }
    c00 = wmma_bf16(a0, b0, c00);
    c01 = wmma_bf16(a0, b1, c01);
    c10 = wmma_bf16(a1, b0, c10);
    c11 = wmma_bf16(a1, b1, c11);
    if (more) { a0 = na0; a1 = na1; b0 = nb0; b1 = nb1; }
  }

  if (outF) {
#pragma unroll
    for (int i = 0; i < 2; ++i) {
      v8f ci0 = i ? c10 : c00;
      v8f ci1 = i ? c11 : c01;
#pragma unroll
      for (int r = 0; r < 8; ++r) {
        const size_t row = (size_t)(tr * 32 + i * 16 + rbase + r);
        outF[row * N + tn * 32 + col] = ci0[r];
        outF[row * N + tn * 32 + 16 + col] = ci1[r];
      }
    }
  } else {
#pragma unroll
    for (int i = 0; i < 2; ++i) {
#pragma unroll
      for (int j = 0; j < 2; ++j) {
        v8f cc = i ? (j ? c11 : c10) : (j ? c01 : c00);
        const int n = tn * 32 + j * 16 + col, h = n >> 6, d = n & 63;
#pragma unroll
        for (int r = 0; r < 8; ++r) {
          const int row = tr * 32 + i * 16 + rbase + r;
          const int l = row / Bdim, bb = row % Bdim;
          outH[(((size_t)bb * HC + h) * Ldim + l) * DPHC + d] = f2bf(cc[r]);
        }
      }
    }
  }
}

// ---------------------------------------------------------------- dir 1->2 stats
// For each (b,h,k): m = max_q S[q,k], Z = sum_q exp(S[q,k]-m). One wave per 16-k tile.
__global__ void __launch_bounds__(128)
colstats(const bf16_t* __restrict__ k1h, const bf16_t* __restrict__ k2h,
         float* __restrict__ cm, float* __restrict__ cz) {
  const int lane = threadIdx.x & 31;
  const int tile = blockIdx.x * 4 + (threadIdx.x >> 5);
  const int KT = L1C >> 4;
  const int bh = tile / KT, kt = tile % KT;
  const bf16_t* k1 = k1h + (size_t)bh * L1C * DPHC;
  const bf16_t* k2 = k2h + (size_t)bh * L2C * DPHC;

  const v16bf b0 = load_bT(k1 + (size_t)(kt * 16) * DPHC + 0, DPHC, lane);
  const v16bf b1 = load_bT(k1 + (size_t)(kt * 16) * DPHC + 32, DPHC, lane);

  float m = -1e30f, Z = 0.f;
  for (int q0 = 0; q0 < L2C; q0 += 16) {
    const bf16_t* arow = k2 + (size_t)q0 * DPHC;
    v16bf a0 = load_a(arow + 0, DPHC, lane);
    v16bf a1 = load_a(arow + 32, DPHC, lane);
    v8f c = zerov8();
    c = wmma_bf16(a0, b0, c);
    c = wmma_bf16(a1, b1, c);
#pragma unroll
    for (int r = 0; r < 8; ++r) {     // online max/sumexp over this column's 8 rows
      const float s = c[r];
      const float nm = fmaxf(m, s);
      Z = Z * __expf(m - nm) + __expf(s - nm);
      m = nm;
    }
  }
  // merge lane l with lane l+16 (same column, other 8 rows)
  const float mo = __shfl_xor(m, 16, 32);
  const float zo = __shfl_xor(Z, 16, 32);
  const float nm = fmaxf(m, mo);
  Z = Z * __expf(m - nm) + zo * __expf(mo - nm);
  m = nm;
  if (lane < 16) {
    const size_t idx = (size_t)bh * L1C + kt * 16 + lane;
    cm[idx] = m;
    cz[idx] = Z;
  }
}

// ---------------------------------------------------------------- dir 2->1 stats
// For each (b,q,k): m = max_h S[b,h,q,k], Z = sum_h exp(S-m). One wave per 16x16 (q,k) tile.
__global__ void __launch_bounds__(128)
headstats(const bf16_t* __restrict__ k1h, const bf16_t* __restrict__ k2h,
          float* __restrict__ hm, float* __restrict__ hz) {
  const int lane = threadIdx.x & 31;
  const int tile = blockIdx.x * 4 + (threadIdx.x >> 5);
  const int KT = L1C >> 4, QT = L2C >> 4;
  const int b = tile / (QT * KT);
  const int qt = (tile / KT) % QT;
  const int kt = tile % KT;
  const int col = lane & 15, rbase = (lane >> 4) * 8;

  float m8[8], z8[8];
#pragma unroll
  for (int r = 0; r < 8; ++r) { m8[r] = -1e30f; z8[r] = 0.f; }

  for (int h = 0; h < HC; ++h) {
    const bf16_t* k1 = k1h + ((size_t)b * HC + h) * L1C * DPHC;
    const bf16_t* k2 = k2h + ((size_t)b * HC + h) * L2C * DPHC;
    v16bf a0 = load_a(k2 + (size_t)(qt * 16) * DPHC + 0, DPHC, lane);
    v16bf a1 = load_a(k2 + (size_t)(qt * 16) * DPHC + 32, DPHC, lane);
    v16bf b0 = load_bT(k1 + (size_t)(kt * 16) * DPHC + 0, DPHC, lane);
    v16bf b1 = load_bT(k1 + (size_t)(kt * 16) * DPHC + 32, DPHC, lane);
    v8f c = zerov8();
    c = wmma_bf16(a0, b0, c);
    c = wmma_bf16(a1, b1, c);
#pragma unroll
    for (int r = 0; r < 8; ++r) {
      const float s = c[r];
      const float nm = fmaxf(m8[r], s);
      z8[r] = z8[r] * __expf(m8[r] - nm) + __expf(s - nm);
      m8[r] = nm;
    }
  }
  const size_t base = ((size_t)b * L2C + qt * 16 + rbase) * L1C + kt * 16 + col;
#pragma unroll
  for (int r = 0; r < 8; ++r) {
    hm[base + (size_t)r * L1C] = m8[r];
    hz[base + (size_t)r * L1C] = z8[r];
  }
}

// ---------------------------------------------------------------- dir 1->2 AV
// O[q,d] = sum_k exp(S[q,k]-cm[k])/cz[k] * v1[k,d], per (b,h,q-tile). LDS relayout C->A frag.
__global__ void __launch_bounds__(128)
ctx12_av(const bf16_t* __restrict__ k1h, const bf16_t* __restrict__ k2h,
         const bf16_t* __restrict__ v1h, const float* __restrict__ cm,
         const float* __restrict__ cz, bf16_t* __restrict__ out /* [L2,B,M] bf16 */) {
  __shared__ __align__(16) bf16_t Plds[4][16 * PSTR];
  const int lane = threadIdx.x & 31;
  const int wave = threadIdx.x >> 5;
  bf16_t* P = Plds[wave];
  const int tile = blockIdx.x * 4 + wave;
  const int QT = L2C >> 4;
  const int bh = tile / QT, qt = tile % QT;
  const int b = bh / HC, h = bh % HC;
  const int col = lane & 15, rbase = (lane >> 4) * 8;

  const bf16_t* k1 = k1h + (size_t)bh * L1C * DPHC;
  const bf16_t* k2 = k2h + (size_t)bh * L2C * DPHC;
  const bf16_t* v1 = v1h + (size_t)bh * L1C * DPHC;

  const v16bf sa0 = load_a(k2 + (size_t)(qt * 16) * DPHC + 0, DPHC, lane);
  const v16bf sa1 = load_a(k2 + (size_t)(qt * 16) * DPHC + 32, DPHC, lane);

  v8f O[4];
#pragma unroll
  for (int dt = 0; dt < 4; ++dt) O[dt] = zerov8();

  for (int k0 = 0; k0 < L1C; k0 += 32) {
#pragma unroll
    for (int sub = 0; sub < 2; ++sub) {
      const int kt16 = k0 + sub * 16;
      v16bf b0 = load_bT(k1 + (size_t)kt16 * DPHC + 0, DPHC, lane);
      v16bf b1 = load_bT(k1 + (size_t)kt16 * DPHC + 32, DPHC, lane);
      v8f c = zerov8();
      c = wmma_bf16(sa0, b0, c);
      c = wmma_bf16(sa1, b1, c);
      const float mc = cm[(size_t)bh * L1C + kt16 + col];
      const float rz = 1.0f / cz[(size_t)bh * L1C + kt16 + col];
#pragma unroll
      for (int r = 0; r < 8; ++r)
        P[(rbase + r) * PSTR + sub * 16 + col] = f2bf(__expf(c[r] - mc) * rz);
    }
    asm volatile("s_wait_dscnt 0" ::: "memory");  // wave-local LDS hand-off
    v16bf pa;  // A-frag of P (16q x 32k)
    {
      const int row = lane & 15, kb = (lane >> 4) * 16;
#pragma unroll
      for (int i = 0; i < 8; ++i) {
        pa[i] = P[row * PSTR + kb + i];
        pa[8 + i] = P[row * PSTR + kb + 8 + i];
      }
    }
#pragma unroll
    for (int dt = 0; dt < 4; ++dt) {
      v16bf vb;  // B-frag: B[k][d] = v1[k][d], k = k0 + kb + i
      const int d = dt * 16 + col, kb = (lane >> 4) * 16;
#pragma unroll
      for (int i = 0; i < 16; ++i) vb[i] = v1[(size_t)(k0 + kb + i) * DPHC + d];
      O[dt] = wmma_bf16(pa, vb, O[dt]);
    }
  }
#pragma unroll
  for (int dt = 0; dt < 4; ++dt) {
    const int mcol = h * DPHC + dt * 16 + col;
#pragma unroll
    for (int r = 0; r < 8; ++r) {
      const int q = qt * 16 + rbase + r;
      out[((size_t)q * BC + b) * MC + mcol] = f2bf(O[dt][r]);
    }
  }
}

// ---------------------------------------------------------------- dir 2->1 AV
// O[k,d] = sum_q exp(S[q,k]-hm[b,q,k])/hz[b,q,k] * v2[q,d], per (b,h,k-tile).
// P tile transposed through LDS to feed P^T as the WMMA A operand.
__global__ void __launch_bounds__(128)
ctx21_av(const bf16_t* __restrict__ k1h, const bf16_t* __restrict__ k2h,
         const bf16_t* __restrict__ v2h, const float* __restrict__ hm,
         const float* __restrict__ hz, bf16_t* __restrict__ out /* [L1,B,M] bf16 */) {
  __shared__ __align__(16) bf16_t Plds[4][16 * PSTR];
  const int lane = threadIdx.x & 31;
  const int wave = threadIdx.x >> 5;
  bf16_t* PT = Plds[wave];
  const int tile = blockIdx.x * 4 + wave;
  const int KT = L1C >> 4;
  const int bh = tile / KT, kt = tile % KT;
  const int b = bh / HC, h = bh % HC;
  const int col = lane & 15, rbase = (lane >> 4) * 8;

  const bf16_t* k1 = k1h + (size_t)bh * L1C * DPHC;
  const bf16_t* k2 = k2h + (size_t)bh * L2C * DPHC;
  const bf16_t* v2 = v2h + (size_t)bh * L2C * DPHC;

  const v16bf sb0 = load_bT(k1 + (size_t)(kt * 16) * DPHC + 0, DPHC, lane);
  const v16bf sb1 = load_bT(k1 + (size_t)(kt * 16) * DPHC + 32, DPHC, lane);

  v8f O[4];
#pragma unroll
  for (int dt = 0; dt < 4; ++dt) O[dt] = zerov8();

  for (int q0 = 0; q0 < L2C; q0 += 32) {
#pragma unroll
    for (int sub = 0; sub < 2; ++sub) {
      const int qt16 = q0 + sub * 16;
      v16bf a0 = load_a(k2 + (size_t)qt16 * DPHC + 0, DPHC, lane);
      v16bf a1 = load_a(k2 + (size_t)qt16 * DPHC + 32, DPHC, lane);
      v8f c = zerov8();
      c = wmma_bf16(a0, sb0, c);
      c = wmma_bf16(a1, sb1, c);
      const size_t sbase = ((size_t)b * L2C + qt16 + rbase) * L1C + kt * 16 + col;
      v8bf pv;
#pragma unroll
      for (int r = 0; r < 8; ++r) {
        const float mm = hm[sbase + (size_t)r * L1C];
        const float zz = hz[sbase + (size_t)r * L1C];
        pv[r] = f2bf(__expf(c[r] - mm) / zz);
      }
      // transposed store: PT[k=col][q = sub*16 + rbase + r], 8 contiguous -> ds_store_b128
      *(v8bf*)&PT[col * PSTR + sub * 16 + rbase] = pv;
    }
    asm volatile("s_wait_dscnt 0" ::: "memory");
    v16bf pa;  // A-frag of P^T (16k x 32q)
    {
      const int row = lane & 15, kb = (lane >> 4) * 16;
#pragma unroll
      for (int i = 0; i < 8; ++i) {
        pa[i] = PT[row * PSTR + kb + i];
        pa[8 + i] = PT[row * PSTR + kb + 8 + i];
      }
    }
#pragma unroll
    for (int dt = 0; dt < 4; ++dt) {
      v16bf vb;  // B-frag: B[q][d] = v2[q][d], q = q0 + kb + i
      const int d = dt * 16 + col, kb = (lane >> 4) * 16;
#pragma unroll
      for (int i = 0; i < 16; ++i) vb[i] = v2[(size_t)(q0 + kb + i) * DPHC + d];
      O[dt] = wmma_bf16(pa, vb, O[dt]);
    }
  }
#pragma unroll
  for (int dt = 0; dt < 4; ++dt) {
    const int mcol = h * DPHC + dt * 16 + col;
#pragma unroll
    for (int r = 0; r < 8; ++r) {
      const int k = kt * 16 + rbase + r;
      out[((size_t)k * BC + b) * MC + mcol] = f2bf(O[dt][r]);
    }
  }
}

// ---------------------------------------------------------------- launch
extern "C" void kernel_launch(void* const* d_in, const int* in_sizes, int n_in,
                              void* d_out, int out_size, void* d_ws, size_t ws_size,
                              hipStream_t stream) {
  (void)in_sizes; (void)n_in; (void)out_size; (void)ws_size;
  const float* ctx1 = (const float*)d_in[0];
  const float* ctx2 = (const float*)d_in[1];
  const float* Wk1 = (const float*)d_in[2];
  const float* bk1 = (const float*)d_in[3];
  const float* Wv1 = (const float*)d_in[4];
  const float* bv1 = (const float*)d_in[5];
  const float* Wk2 = (const float*)d_in[6];
  const float* bk2 = (const float*)d_in[7];
  const float* Wv2 = (const float*)d_in[8];
  const float* bv2 = (const float*)d_in[9];
  const float* Wf12 = (const float*)d_in[10];
  const float* bf12 = (const float*)d_in[11];
  const float* Wf21 = (const float*)d_in[12];
  const float* bf21 = (const float*)d_in[13];

  float* out21 = (float*)d_out;                  // context_2_to_1 [L1,B,M]
  float* out12 = out21 + (size_t)L1C * BC * MC;  // context_1_to_2 [L2,B,M]

  uint8_t* ws = (uint8_t*)d_ws;
  size_t off = 0;
  auto carve = [&](size_t bytes) -> void* {
    void* p = ws + off;
    off += (bytes + 255) & ~(size_t)255;
    return p;
  };
  bf16_t* ctx1h = (bf16_t*)carve((size_t)L1C * BC * D1C * 2);
  bf16_t* ctx2h = (bf16_t*)carve((size_t)L2C * BC * D2C * 2);
  bf16_t* Wk1h = (bf16_t*)carve((size_t)MC * D1C * 2);
  bf16_t* Wv1h = (bf16_t*)carve((size_t)MC * D1C * 2);
  bf16_t* Wk2h = (bf16_t*)carve((size_t)MC * D2C * 2);
  bf16_t* Wv2h = (bf16_t*)carve((size_t)MC * D2C * 2);
  bf16_t* Wf12h = (bf16_t*)carve((size_t)MC * MC * 2);
  bf16_t* Wf21h = (bf16_t*)carve((size_t)MC * MC * 2);
  bf16_t* k1hb = (bf16_t*)carve((size_t)BC * HC * L1C * DPHC * 2);
  bf16_t* v1hb = (bf16_t*)carve((size_t)BC * HC * L1C * DPHC * 2);
  bf16_t* k2hb = (bf16_t*)carve((size_t)BC * HC * L2C * DPHC * 2);
  bf16_t* v2hb = (bf16_t*)carve((size_t)BC * HC * L2C * DPHC * 2);
  float* cmb = (float*)carve((size_t)BC * HC * L1C * 4);
  float* czb = (float*)carve((size_t)BC * HC * L1C * 4);
  float* hmb = (float*)carve((size_t)BC * L2C * L1C * 4);
  float* hzb = (float*)carve((size_t)BC * L2C * L1C * 4);
  bf16_t* c12h = (bf16_t*)carve((size_t)L2C * BC * MC * 2);
  bf16_t* c21h = (bf16_t*)carve((size_t)L1C * BC * MC * 2);

  auto cvt = [&](const float* s, bf16_t* d, size_t n) {
    const int blocks = (int)((n + 256 * 8 - 1) / (256 * 8));
    cvt_f32_bf16<<<blocks, 256, 0, stream>>>(s, d, (int)n);
  };
  cvt(ctx1, ctx1h, (size_t)L1C * BC * D1C);
  cvt(ctx2, ctx2h, (size_t)L2C * BC * D2C);
  cvt(Wk1, Wk1h, (size_t)MC * D1C);
  cvt(Wv1, Wv1h, (size_t)MC * D1C);
  cvt(Wk2, Wk2h, (size_t)MC * D2C);
  cvt(Wv2, Wv2h, (size_t)MC * D2C);
  cvt(Wf12, Wf12h, (size_t)MC * MC);
  cvt(Wf21, Wf21h, (size_t)MC * MC);

  const int R = L1C * BC;                             // 8192 rows
  const int projBlocks = (R / 32) * (MC / 32) / 4;    // 32x32 per wave, 4 waves/block

  // input projections -> [B,H,L,64] bf16
  proj_gemm<<<projBlocks, 128, 0, stream>>>(ctx1h, Wk1h, bk1, D1C, MC, BC, L1C, nullptr, k1hb);
  proj_gemm<<<projBlocks, 128, 0, stream>>>(ctx1h, Wv1h, bv1, D1C, MC, BC, L1C, nullptr, v1hb);
  proj_gemm<<<projBlocks, 128, 0, stream>>>(ctx2h, Wk2h, bk2, D2C, MC, BC, L2C, nullptr, k2hb);
  proj_gemm<<<projBlocks, 128, 0, stream>>>(ctx2h, Wv2h, bv2, D2C, MC, BC, L2C, nullptr, v2hb);

  colstats<<<BC * HC * (L1C / 16) / 4, 128, 0, stream>>>(k1hb, k2hb, cmb, czb);
  headstats<<<BC * (L2C / 16) * (L1C / 16) / 4, 128, 0, stream>>>(k1hb, k2hb, hmb, hzb);

  ctx12_av<<<BC * HC * (L2C / 16) / 4, 128, 0, stream>>>(k1hb, k2hb, v1hb, cmb, czb, c12h);
  ctx21_av<<<BC * HC * (L1C / 16) / 4, 128, 0, stream>>>(k1hb, k2hb, v2hb, hmb, hzb, c21h);

  // final projections, fp32 straight into d_out ([L,B,M] row order == GEMM row order)
  proj_gemm<<<projBlocks, 128, 0, stream>>>(c21h, Wf21h, bf21, MC, MC, BC, L1C, out21, nullptr);
  proj_gemm<<<projBlocks, 128, 0, stream>>>(c12h, Wf12h, bf12, MC, MC, BC, L2C, out12, nullptr);
}
